// QuantConv2d_24678882082820
// MI455X (gfx1250) — compile-verified, matrix operand order checked by reading, not run on hardware
//
#include <hip/hip_runtime.h>
#include <math.h>

typedef __attribute__((ext_vector_type(16))) _Float16 v16h;
typedef __attribute__((ext_vector_type(8)))  _Float16 v8h;
typedef __attribute__((ext_vector_type(4)))  _Float16 v4h;
typedef __attribute__((ext_vector_type(8)))  float    v8f;
typedef __attribute__((ext_vector_type(4)))  unsigned int u32x4;
typedef __attribute__((ext_vector_type(4)))  int i32x4;
typedef __attribute__((ext_vector_type(8)))  int i32x8;

#define IMG   32
#define CIN   128
#define COUT  256
#define HW    56
#define SP    (HW * HW)          // 3136
#define LDSP  40                 // padded LDS row (halves): 32 data + 8 pad, 80B
#define KSTEPS 36                // 9 taps * 4 ci-chunks of 32

static constexpr size_t XQ_ELEMS    = (size_t)IMG * SP * CIN;  // NHWC fp16 halves
static constexpr size_t WT_HALF_OFF = XQ_ELEMS;

#if defined(__has_builtin)
#  if __has_builtin(__builtin_amdgcn_tensor_load_to_lds)
#    define HAVE_TDM 1
#  else
#    define HAVE_TDM 0
#  endif
#else
#  define HAVE_TDM 0
#endif

#if defined(__has_builtin) && __has_builtin(__builtin_amdgcn_s_wait_tensorcnt)
#  define WAIT_TENSOR() __builtin_amdgcn_s_wait_tensorcnt(0)
#else
#  define WAIT_TENSOR() asm volatile("s_wait_tensorcnt 0x0" ::: "memory")
#endif

// ---- Kernel 1: quantize fp32 NCHW -> fp16 NHWC via LDS transpose (both sides coalesced) ----
__global__ __launch_bounds__(256) void quant_kernel(const float* __restrict__ x,
                                                    _Float16* __restrict__ xq) {
    __shared__ _Float16 t[64][132];
    const int n   = blockIdx.y;
    const int sp0 = blockIdx.x * 64;
    const size_t nb = (size_t)n * CIN * SP;
#pragma unroll
    for (int it = 0; it < 32; ++it) {
        int l   = threadIdx.x + it * 256;
        int ci  = l >> 6;
        int spi = l & 63;
        float v = x[nb + (size_t)ci * SP + sp0 + spi];
        t[spi][ci] = (_Float16)(rintf(v * 256.0f) * 0.00390625f);
    }
    __syncthreads();
#pragma unroll
    for (int it = 0; it < 8; ++it) {
        int l   = threadIdx.x + it * 256;
        int spi = l >> 5;
        int c4  = l & 31;
        v4h vv = *(const v4h*)&t[spi][c4 * 4];
        *(v4h*)(xq + ((size_t)(n * SP + sp0 + spi)) * CIN + c4 * 4) = vv;
    }
}

// ---- Kernel 2: weight transform 2^w - 2^-w, relayout to [tap][co][ci] fp16 ----
__global__ __launch_bounds__(256) void wprep_kernel(const float* __restrict__ w,
                                                    _Float16* __restrict__ wt) {
    int g   = blockIdx.x * 256 + threadIdx.x;
    int co  = g / (CIN * 9);
    int rem = g - co * (CIN * 9);
    int ci  = rem / 9;
    int k   = rem - ci * 9;
    float wv = w[g];
    wt[(size_t)k * COUT * CIN + (size_t)co * CIN + ci] = (_Float16)(exp2f(wv) - exp2f(-wv));
}

// ---- Kernel 3: implicit-GEMM conv, double-buffered LDS pipeline, TDM for B tiles ----
__global__ __launch_bounds__(256) void conv_kernel(const _Float16* __restrict__ xq,
                                                   const _Float16* __restrict__ wt,
                                                   float* __restrict__ out) {
    __shared__ _Float16 Alds[2][128 * LDSP];
    __shared__ _Float16 Blds[2][128 * LDSP];

    const int tid  = threadIdx.x;
    const int lane = tid & 31;
    const int wid  = tid >> 5;
    const int wm   = wid & 3;
    const int wn   = wid >> 2;
    const int tile_n = blockIdx.x;
    const int Pbase  = blockIdx.y * 128;

    const int sub = tid & 3;
    const int rA0 = tid >> 2;                 // stages rows rA0 and rA0+64

    // ---- prologue: loop-invariant A addressing (no divisions in the hot loop) ----
    int boff[2];                              // byte offset of (pixel, ci=sub*8) in NHWC xq
    unsigned vmask[2] = {0u, 0u};             // 9-bit per-tap validity
#pragma unroll
    for (int j = 0; j < 2; ++j) {
        int P    = Pbase + rA0 + j * 64;
        int nimg = P / SP;
        int spv  = P - nimg * SP;
        int ph   = spv / HW;
        int pw   = spv - ph * HW;
        boff[j]  = (int)(((unsigned)(nimg * SP + ph * HW + pw) * CIN + sub * 8u) * 2u);
#pragma unroll
        for (int tap = 0; tap < 9; ++tap) {
            int ih = ph + tap / 3 - 1, iw = pw + tap % 3 - 1;
            if (((unsigned)ih < HW) && ((unsigned)iw < HW)) vmask[j] |= 1u << tap;
        }
    }
    const _Float16* wbase = wt + (size_t)tile_n * 128 * CIN;

    v8f acc[2][4];
#pragma unroll
    for (int f = 0; f < 2; ++f)
#pragma unroll
        for (int g = 0; g < 4; ++g) acc[f][g] = (v8f)0.0f;

    const int lrow = lane & 15;
    const int ko   = ((lane >> 4) & 1) * 8;

    v8h ra[2];
    auto loadA = [&](int s) {
        int tap = s >> 2, kc = s & 3;         // uniform -> SALU decode
        int delta = ((tap / 3 - 1) * HW + (tap % 3 - 1)) * (CIN * 2) + kc * 64;
#pragma unroll
        for (int j = 0; j < 2; ++j) {
            v8h r = (v8h)(_Float16)0.0f;
            if ((vmask[j] >> tap) & 1u)
                r = *(const v8h*)((const char*)xq + (boff[j] + delta));
            ra[j] = r;
        }
    };
    auto storeA = [&](int b) {
#pragma unroll
        for (int j = 0; j < 2; ++j)
            *(v8h*)(&Alds[b][(rA0 + j * 64) * LDSP + sub * 8]) = ra[j];
    };

#if HAVE_TDM
    auto tdmB = [&](int s, int b) {
        int tap = s >> 2, kc = s & 3;
        const _Float16* g = wbase + (size_t)tap * COUT * CIN + kc * 32;
        unsigned long long ga = (unsigned long long)(uintptr_t)g;
        unsigned lo = (unsigned)(uintptr_t)&Blds[b][0];        // flat low-32 = LDS byte offset
        u32x4 g0;
        g0[0] = 1u;                                            // count=1
        g0[1] = lo;                                            // lds_addr
        g0[2] = (unsigned)(ga & 0xffffffffu);                  // global_addr[31:0]
        g0[3] = (unsigned)((ga >> 32) & 0x1ffffffu) | (2u << 30);  // ga[56:32] | type=2
        i32x8 g1;
        g1[0] = (1 << 16) | (1 << 20) | (3 << 22) | (3 << 25); // 2B, pad_en, 16dw interval, 4dw pad
        g1[1] = 0;
        g1[2] = 0x10;                                          // tensor_dim0 = 1<<20 (no OOB)
        g1[3] = (32 << 16) | 0x10;                             // tile_dim0=32 | tensor_dim1 hi
        g1[4] = 128;                                           // tile_dim1=128
        g1[5] = CIN;                                           // tensor_dim0_stride=128
        g1[6] = 0;
        g1[7] = 0;
        i32x4 gz = (i32x4)0;
#if __clang_major__ >= 23
        i32x8 gz8 = (i32x8)0;
        __builtin_amdgcn_tensor_load_to_lds(g0, g1, gz, gz, gz8, 0);
#else
        __builtin_amdgcn_tensor_load_to_lds(g0, g1, gz, gz, 0);
#endif
    };
#else
    v8h rb[2];
    auto loadB = [&](int s) {
        int tap = s >> 2, kc = s & 3;
#pragma unroll
        for (int j = 0; j < 2; ++j)
            rb[j] = *(const v8h*)(wbase + (size_t)tap * COUT * CIN +
                                  (size_t)(rA0 + j * 64) * CIN + kc * 32 + sub * 8);
    };
    auto storeB = [&](int b) {
#pragma unroll
        for (int j = 0; j < 2; ++j)
            *(v8h*)(&Blds[b][(rA0 + j * 64) * LDSP + sub * 8]) = rb[j];
    };
#endif

    // ---- pipeline prologue ----
    loadA(0);
    storeA(0);
#if HAVE_TDM
    if (wid == 0) { tdmB(0, 0); WAIT_TENSOR(); }
#else
    loadB(0); storeB(0);
#endif
    __syncthreads();

    int buf = 0;
    for (int s = 0; s < KSTEPS; ++s) {
        const bool more = (s + 1) < KSTEPS;
        if (more) {
            loadA(s + 1);
#if HAVE_TDM
            if (wid == 0) tdmB(s + 1, buf ^ 1);
#else
            loadB(s + 1);
#endif
        }
        union { v16h v; v8h h[2]; } afr[2], bfr[4];
#pragma unroll
        for (int f = 0; f < 2; ++f) {
            int r = wm * 32 + f * 16 + lrow;
            afr[f].h[0] = *(const v8h*)(&Alds[buf][r * LDSP + ko]);
            afr[f].h[1] = *(const v8h*)(&Alds[buf][r * LDSP + 16 + ko]);
        }
#pragma unroll
        for (int g = 0; g < 4; ++g) {
            int c = wn * 64 + g * 16 + lrow;
            bfr[g].h[0] = *(const v8h*)(&Blds[buf][c * LDSP + ko]);
            bfr[g].h[1] = *(const v8h*)(&Blds[buf][c * LDSP + 16 + ko]);
        }
#pragma unroll
        for (int f = 0; f < 2; ++f)
#pragma unroll
            for (int g = 0; g < 4; ++g)
                acc[f][g] = __builtin_amdgcn_wmma_f32_16x16x32_f16(
                    false, afr[f].v, false, bfr[g].v, (short)0, acc[f][g], false, false);
        if (more) {
            storeA(buf ^ 1);
#if HAVE_TDM
            if (wid == 0) WAIT_TENSOR();
#else
            storeB(buf ^ 1);
#endif
        }
        __syncthreads();
        buf ^= 1;
    }

    // ---- epilogue: f32 C/D layout -> NCHW stores ----
    const int nloc = lane & 15;
    const int mofs = (lane >> 4) * 8;
#pragma unroll
    for (int f = 0; f < 2; ++f) {
#pragma unroll
        for (int g = 0; g < 4; ++g) {
            int co = tile_n * 128 + wn * 64 + g * 16 + nloc;
#pragma unroll
            for (int i = 0; i < 8; ++i) {
                int m  = mofs + i;
                int Pm = Pbase + wm * 32 + f * 16 + m;
                int n2 = Pm / SP;
                int s2 = Pm - n2 * SP;
                out[(size_t)n2 * COUT * SP + (size_t)co * SP + s2] = acc[f][g][i];
            }
        }
    }
}

extern "C" void kernel_launch(void* const* d_in, const int* in_sizes, int n_in,
                              void* d_out, int out_size, void* d_ws, size_t ws_size,
                              hipStream_t stream) {
    const float* x = (const float*)d_in[0];   // [32,128,56,56]
    const float* w = (const float*)d_in[1];   // [256,128,3,3]
    float* out = (float*)d_out;               // [32,256,56,56]

    _Float16* xq = (_Float16*)d_ws;                    // NHWC fp16
    _Float16* wt = (_Float16*)d_ws + WT_HALF_OFF;      // [9][256][128] fp16

    quant_kernel<<<dim3(49, 32), 256, 0, stream>>>(x, xq);
    wprep_kernel<<<1152, 256, 0, stream>>>(w, wt);
    conv_kernel<<<dim3(2, 784), 256, 0, stream>>>(xq, wt, out);
}
